// Net_77541339562219
// MI455X (gfx1250) — compile-verified, matrix operand order checked by reading, not run on hardware
//
#include <hip/hip_runtime.h>
#include <hip/hip_bf16.h>
#include <math.h>

// ---------------------------------------------------------------------------
// CDNA5 (gfx1250) WMMA types
// ---------------------------------------------------------------------------
typedef __attribute__((ext_vector_type(16))) __bf16 v16bf;
typedef __attribute__((ext_vector_type(8)))  __bf16 v8bf;
typedef __attribute__((ext_vector_type(8)))  float  v8f;

__device__ __forceinline__ __bf16 f2bf(float f) {
  unsigned int u = __builtin_bit_cast(unsigned int, f);
  unsigned int r = u + 0x7FFFu + ((u >> 16) & 1u);   // round-to-nearest-even
  unsigned short h = (unsigned short)(r >> 16);
  return __builtin_bit_cast(__bf16, h);
}
__device__ __forceinline__ float bf2f(__bf16 b) {
  unsigned int u = ((unsigned int)__builtin_bit_cast(unsigned short, b)) << 16;
  return __builtin_bit_cast(float, u);
}
__device__ __forceinline__ float sigf(float x) { return 1.0f / (1.0f + __expf(-x)); }

// Concatenate two contiguous 8-element bf16 vectors into a WMMA fragment.
__device__ __forceinline__ v16bf cat16(v8bf lo, v8bf hi) {
  return __builtin_shufflevector(lo, hi, 0, 1, 2, 3, 4, 5, 6, 7,
                                 8, 9, 10, 11, 12, 13, 14, 15);
}

// ---------------------------------------------------------------------------
// Elementwise: f32 -> bf16 convert
// ---------------------------------------------------------------------------
__global__ void k_f32_to_bf16(const float* __restrict__ in, __bf16* __restrict__ out, long n) {
  long i = (long)blockIdx.x * blockDim.x + threadIdx.x;
  if (i < n) out[i] = f2bf(in[i]);
}

// Weight pre-pass: f32 [M,Kreal] -> bf16 [M,Kpad], zero-padded rows.
// Gives 16B-aligned contiguous K-runs so A fragments load as global_load_b128.
__global__ void k_wpad_bf16(const float* __restrict__ in, __bf16* __restrict__ out,
                            int M, int Kreal, int Kpad) {
  long i = (long)blockIdx.x * blockDim.x + threadIdx.x;
  long n = (long)M * Kpad;
  if (i >= n) return;
  int k = (int)(i % Kpad);
  long m = i / Kpad;
  out[i] = (k < Kreal) ? f2bf(in[m * Kreal + k]) : f2bf(0.0f);
}

// ---------------------------------------------------------------------------
// 2x2 max-pool, stride 2, on bf16 NCHW planes
// ---------------------------------------------------------------------------
__global__ void k_maxpool2_bf16(const __bf16* __restrict__ in, __bf16* __restrict__ out,
                                int BC, int H, int W) {
  const int OH = H >> 1, OW = W >> 1;
  const long n = (long)BC * OH * OW;
  long i = (long)blockIdx.x * blockDim.x + threadIdx.x;
  if (i >= n) return;
  int ow = (int)(i % OW); long t = i / OW;
  int oh = (int)(t % OH); int p = (int)(t / OH);
  const __bf16* base = in + ((long)p * H + (long)oh * 2) * W + (long)ow * 2;
  float a = bf2f(base[0]), b = bf2f(base[1]);
  float c = bf2f(base[W]), d = bf2f(base[W + 1]);
  out[i] = f2bf(fmaxf(fmaxf(a, b), fmaxf(c, d)));
}

// ---------------------------------------------------------------------------
// Implicit-GEMM convolution via v_wmma_f32_16x16x32_bf16.
//   C[co, n=(b,oy,ox)] = sum_k Wbf[co,k] * in[b,ci,oy+ky,ox+kx],  k=(ci,ky,kx)
// Per wave: MBLK 16x16 tiles sharing one gathered B fragment.
// A fragments: two global_load_b128 from bf16-padded weights.
// B gather:   LDS offset table off[k] = ci*H*W + ky*W + kx (built once per
//             block, -1 = K padding) kills the per-element div/mod chains.
// Fragment layouts per CDNA5 ISA 7.12.2; wave-uniform control flow
// keeps EXEC all-ones at every WMMA.
// ---------------------------------------------------------------------------
#define MAX_KPAD 3200

template <int MBLK>
__global__ __launch_bounds__(256) void k_conv_wmma(
    const __bf16* __restrict__ in, const __bf16* __restrict__ wbf,
    const float* __restrict__ bias, __bf16* __restrict__ out,
    int Bn, int Cin, int H, int W, int Cout,
    int KH, int KW, int OH, int OW, int Kreal, int Kpad)
{
  __shared__ int offs[MAX_KPAD];
  const int HW = H * W;
  const int KHW = KH * KW;
  // Cooperative offset-table build (before any divergence between waves).
  for (int k = threadIdx.x; k < Kpad; k += blockDim.x) {
    if (k < Kreal) {
      const int ci = k / KHW;
      const int kr = k - ci * KHW;
      const int ky = kr / KW;
      const int kx = kr - ky * KW;
      offs[k] = ci * HW + ky * W + kx;
    } else {
      offs[k] = -1;
    }
  }
  __syncthreads();

  const int wave = threadIdx.x >> 5;          // 8 waves / block, one tile-group each
  const int lane = threadIdx.x & 31;          // wave32
  const long Ntot   = (long)Bn * OH * OW;
  const long ntiles = (Ntot + 15) >> 4;
  const long ntile  = (long)blockIdx.x * 8 + wave;
  if (ntile >= ntiles) return;                // wave-uniform (after the only barrier)

  const int hl = lane >> 4;                   // K-half select
  const int lo = lane & 15;
  const long n = ntile * 16 + lo;             // B column gathered by this lane
  const bool nvalid = (n < Ntot);
  const int OHW = OH * OW;
  int b = 0, oy = 0, ox = 0;
  if (nvalid) {
    b = (int)(n / OHW);
    int rr = (int)(n % OHW);
    oy = rr / OW; ox = rr - oy * OW;
  }
  const long base_n = (long)b * Cin * HW + (long)oy * W + ox;

  // A row base for each M sub-tile (bf16, padded stride Kpad, 16B aligned runs)
  const __bf16* arow[MBLK];
#pragma unroll
  for (int mb = 0; mb < MBLK; ++mb) {
    const int m = (blockIdx.y * MBLK + mb) * 16 + lo;
    arow[mb] = wbf + (long)m * Kpad;
  }

  v8f acc[MBLK];
#pragma unroll
  for (int mb = 0; mb < MBLK; ++mb) acc[mb] = (v8f){0.f,0.f,0.f,0.f,0.f,0.f,0.f,0.f};

  for (int kc = 0; kc < Kpad; kc += 32) {
    const int k0 = kc + hl * 8;               // contiguous run j=0..7
    const int k1 = k0 + 16;                   // contiguous run j=8..15

    // B fragment: ds_load the offset table in 128-bit chunks, then gather.
    int ko[16];
    *(int4*)&ko[0]  = *(const int4*)&offs[k0];
    *(int4*)&ko[4]  = *(const int4*)&offs[k0 + 4];
    *(int4*)&ko[8]  = *(const int4*)&offs[k1];
    *(int4*)&ko[12] = *(const int4*)&offs[k1 + 4];
    v16bf bfv;
#pragma unroll
    for (int j = 0; j < 16; ++j) {
      const int off = ko[j];
      __bf16 bv = f2bf(0.0f);
      if (nvalid && off >= 0) bv = in[base_n + off];
      bfv[j] = bv;
    }

#pragma unroll
    for (int mb = 0; mb < MBLK; ++mb) {
      __builtin_prefetch(arow[mb] + kc + 32, 0, 1);     // global_prefetch_b8
      const v8bf a_lo = *(const v8bf*)(arow[mb] + k0);  // global_load_b128
      const v8bf a_hi = *(const v8bf*)(arow[mb] + k1);  // global_load_b128
      acc[mb] = __builtin_amdgcn_wmma_f32_16x16x32_bf16(
          false, cat16(a_lo, a_hi), false, bfv, (short)0, acc[mb], false, false);
    }
  }

  if (nvalid) {
#pragma unroll
    for (int mb = 0; mb < MBLK; ++mb) {
#pragma unroll
      for (int v = 0; v < 8; ++v) {
        const int mm = (blockIdx.y * MBLK + mb) * 16 + v + 8 * hl;
        float val = fmaxf(acc[mb][v] + bias[mm], 0.0f);
        out[(((long)b * Cout + mm) * OH + oy) * OW + ox] = f2bf(val);
      }
    }
  }
}

// ---------------------------------------------------------------------------
// FC GEMM: C[m,n] = sum_k Wbf[m,k] * act[n,k]   (both bf16, K % 32 == 0)
// Per wave: one A fragment feeds NBLK=2 WMMAs. All operand loads are b128.
// mode 0: bf16 out, no activation.  mode 1: f32 out, fused sigmoid.
// Output stored row-major [N, M].
// ---------------------------------------------------------------------------
__global__ __launch_bounds__(256) void k_fc_wmma(
    const __bf16* __restrict__ act, const __bf16* __restrict__ wbf,
    const float* __restrict__ bias, __bf16* __restrict__ out_bf,
    float* __restrict__ out_f32, int M, int K, int N, int mode)
{
  const int NBLK = 2;
  const int wave = threadIdx.x >> 5;
  const int lane = threadIdx.x & 31;
  const int ngrp  = (N + 16 * NBLK - 1) / (16 * NBLK);
  const int grp   = blockIdx.x * 8 + wave;
  if (grp >= ngrp) return;                    // wave-uniform exit
  const int hl = lane >> 4;
  const int lo = lane & 15;
  const int m  = blockIdx.y * 16 + lo;
  const __bf16* arow = wbf + (long)m * K;

  const __bf16* brow[NBLK];
  bool nv[NBLK];
  int nn[NBLK];
#pragma unroll
  for (int nb = 0; nb < NBLK; ++nb) {
    nn[nb] = (grp * NBLK + nb) * 16 + lo;
    nv[nb] = nn[nb] < N;
    brow[nb] = act + (long)(nv[nb] ? nn[nb] : 0) * K;
  }

  v8f acc[NBLK];
#pragma unroll
  for (int nb = 0; nb < NBLK; ++nb) acc[nb] = (v8f){0.f,0.f,0.f,0.f,0.f,0.f,0.f,0.f};

  for (int kc = 0; kc < K; kc += 32) {
    const int k0 = kc + hl * 8;
    const int k1 = k0 + 16;
    __builtin_prefetch(arow + kc + 32, 0, 1);
    const v16bf af = cat16(*(const v8bf*)(arow + k0), *(const v8bf*)(arow + k1));
#pragma unroll
    for (int nb = 0; nb < NBLK; ++nb) {
      const v16bf bfv = cat16(*(const v8bf*)(brow[nb] + k0),
                              *(const v8bf*)(brow[nb] + k1));
      acc[nb] = __builtin_amdgcn_wmma_f32_16x16x32_bf16(
          false, af, false, bfv, (short)0, acc[nb], false, false);
    }
  }

#pragma unroll
  for (int nb = 0; nb < NBLK; ++nb) {
    if (!nv[nb]) continue;
#pragma unroll
    for (int v = 0; v < 8; ++v) {
      const int mm = blockIdx.y * 16 + v + 8 * hl;
      float val = acc[nb][v] + bias[mm];
      if (mode == 1) {
        out_f32[(long)nn[nb] * M + mm] = sigf(val);
      } else {
        out_bf[(long)nn[nb] * M + mm] = f2bf(val);
      }
    }
  }
}

// ---------------------------------------------------------------------------
// LSTM scan: 512 serial steps, H=384, gates 1536. Latency-bound by the serial
// dependence; one persistent workgroup (512 thr = 16 waves on one WGP),
// x/h/c/gates in LDS, f32 weights (4.7 MB) resident in the 192 MB L2.
// float4 (b128) operand loads throughout.
// ---------------------------------------------------------------------------
__global__ __launch_bounds__(512) void k_lstm(
    const float* __restrict__ seq,   // [T, 384]
    const float* __restrict__ w_ih,  // [1536, 384]
    const float* __restrict__ w_hh,  // [1536, 384]
    const float* __restrict__ b_ih, const float* __restrict__ b_hh,
    float* __restrict__ out, int T)
{
  __shared__ __align__(16) float xs[384];
  __shared__ __align__(16) float hs[384];
  __shared__ __align__(16) float cs[384];
  __shared__ __align__(16) float gates[1536];
  const int tid = threadIdx.x;
  for (int i = tid; i < 384; i += 512) { hs[i] = 0.f; cs[i] = 0.f; }
  __syncthreads();

  const float4* xs4 = (const float4*)xs;
  const float4* hs4 = (const float4*)hs;

  for (int t = 0; t < T; ++t) {
    for (int i = tid; i < 384; i += 512) xs[i] = seq[(long)t * 384 + i];
    __syncthreads();

#pragma unroll
    for (int s = 0; s < 3; ++s) {              // 512 thr x 3 rows = 1536 gates
      const int row = tid + s * 512;
      const float4* wi4 = (const float4*)(w_ih + (long)row * 384);
      const float4* wh4 = (const float4*)(w_hh + (long)row * 384);
      float a = b_ih[row] + b_hh[row];
#pragma unroll 4
      for (int k = 0; k < 96; ++k) {
        const float4 w = wi4[k], xv = xs4[k];
        a = fmaf(w.x, xv.x, a); a = fmaf(w.y, xv.y, a);
        a = fmaf(w.z, xv.z, a); a = fmaf(w.w, xv.w, a);
      }
#pragma unroll 4
      for (int k = 0; k < 96; ++k) {
        const float4 w = wh4[k], hv = hs4[k];
        a = fmaf(w.x, hv.x, a); a = fmaf(w.y, hv.y, a);
        a = fmaf(w.z, hv.z, a); a = fmaf(w.w, hv.w, a);
      }
      gates[row] = a;
    }
    __syncthreads();

    if (tid < 384) {
      const float ig = sigf(gates[tid]);
      const float fg = sigf(gates[384 + tid]);
      const float gg = tanhf(gates[768 + tid]);
      const float og = sigf(gates[1152 + tid]);
      const float c  = fg * cs[tid] + ig * gg;
      const float h  = og * tanhf(c);
      cs[tid] = c; hs[tid] = h;
      out[(long)t * 384 + tid] = h;
    }
    __syncthreads();
  }
}

// ---------------------------------------------------------------------------
// Orchestration
// ---------------------------------------------------------------------------
static inline unsigned int cdivu(long a, long b) { return (unsigned int)((a + b - 1) / b); }

extern "C" void kernel_launch(void* const* d_in, const int* in_sizes, int n_in,
                              void* d_out, int out_size, void* d_ws, size_t ws_size,
                              hipStream_t stream)
{
  (void)in_sizes; (void)n_in; (void)out_size; (void)ws_size;
  const float* x   = (const float*)d_in[0];
  const float* cw1 = (const float*)d_in[1];  const float* cb1 = (const float*)d_in[2];
  const float* cw2 = (const float*)d_in[3];  const float* cb2 = (const float*)d_in[4];
  const float* cw3 = (const float*)d_in[5];  const float* cb3 = (const float*)d_in[6];
  const float* cw4 = (const float*)d_in[7];  const float* cb4 = (const float*)d_in[8];
  const float* cw5 = (const float*)d_in[9];  const float* cb5 = (const float*)d_in[10];
  const float* cw6 = (const float*)d_in[11]; const float* cb6 = (const float*)d_in[12];
  const float* lw1 = (const float*)d_in[13]; const float* lb1 = (const float*)d_in[14];
  const float* lw2 = (const float*)d_in[15]; const float* lb2 = (const float*)d_in[16];
  const float* wih = (const float*)d_in[17]; const float* whh = (const float*)d_in[18];
  const float* bih = (const float*)d_in[19]; const float* bhh = (const float*)d_in[20];
  float* out = (float*)d_out;

  // Workspace layout (all 16B aligned):
  //   bufA : 72,384,512 bf16 (conv1 out, max)           ~145 MB
  //   bufB : 18,096,128 bf16 (pool1 out, max)           ~36 MB
  //   seq  : 512*384 f32                                ~0.8 MB
  //   wbf  : bf16 K-padded weights, 9,604,096 elems     ~19 MB
  char* ws = (char*)d_ws;
  const size_t szA = 72384512ull * 2;
  const size_t szB = 18096128ull * 2;
  const size_t szS = 512ull * 384 * 4;
  __bf16* bufA = (__bf16*)(ws);
  __bf16* bufB = (__bf16*)(ws + szA);
  float*  seq  = (float*)(ws + szA + szB);
  __bf16* wpool = (__bf16*)(ws + szA + szB + szS);

  __bf16* wb1 = wpool;                 // 16  x 64   = 1024
  __bf16* wb2 = wb1 + 1024;            // 32  x 256  = 8192
  __bf16* wb3 = wb2 + 8192;            // 64  x 800  = 51200
  __bf16* wb4 = wb3 + 51200;           // 128 x 1600 = 204800
  __bf16* wb5 = wb4 + 204800;          // 128 x 3200 = 409600
  __bf16* wb6 = wb5 + 409600;          // 128 x 1152 = 147456
  __bf16* wf1 = wb6 + 147456;          // 1024x 8192 = 8388608
  __bf16* wf2 = wf1 + 8388608;         // 384 x 1024 = 393216

  const dim3 blk(256);

  // Weight pre-pass: f32 -> bf16, K-padded rows.
  { long n; 
    n = 16l*64;      k_wpad_bf16<<<dim3(cdivu(n,256)),blk,0,stream>>>(cw1, wb1, 16, 49, 64);
    n = 32l*256;     k_wpad_bf16<<<dim3(cdivu(n,256)),blk,0,stream>>>(cw2, wb2, 32, 256, 256);
    n = 64l*800;     k_wpad_bf16<<<dim3(cdivu(n,256)),blk,0,stream>>>(cw3, wb3, 64, 800, 800);
    n = 128l*1600;   k_wpad_bf16<<<dim3(cdivu(n,256)),blk,0,stream>>>(cw4, wb4, 128, 1600, 1600);
    n = 128l*3200;   k_wpad_bf16<<<dim3(cdivu(n,256)),blk,0,stream>>>(cw5, wb5, 128, 3200, 3200);
    n = 128l*1152;   k_wpad_bf16<<<dim3(cdivu(n,256)),blk,0,stream>>>(cw6, wb6, 128, 1152, 1152);
    n = 1024l*8192;  k_wpad_bf16<<<dim3(cdivu(n,256)),blk,0,stream>>>(lw1, wf1, 1024, 8192, 8192);
    n = 384l*1024;   k_wpad_bf16<<<dim3(cdivu(n,256)),blk,0,stream>>>(lw2, wf2, 384, 1024, 1024);
  }

  // input f32 -> bf16 (into bufB)
  { long n = 512l * 100 * 100;
    k_f32_to_bf16<<<dim3(cdivu(n, 256)), blk, 0, stream>>>(x, bufB, n); }

  // conv1: 1x100x100 -> 16x94x94 (K=49 pad 64), MBLK=1   bufB -> bufA
  { long Nt = 512l * 94 * 94; dim3 g(cdivu((Nt + 15) >> 4, 8), 1);
    k_conv_wmma<1><<<g, blk, 0, stream>>>(bufB, wb1, cb1, bufA,
        512, 1, 100, 100, 16, 7, 7, 94, 94, 49, 64); }
  // pool1: 94 -> 47                                      bufA -> bufB
  { long n = 512l * 16 * 47 * 47;
    k_maxpool2_bf16<<<dim3(cdivu(n, 256)), blk, 0, stream>>>(bufA, bufB, 512 * 16, 94, 94); }

  // conv2: 16x47x47 -> 32x44x44 (K=256), MBLK=2          bufB -> bufA
  { long Nt = 512l * 44 * 44; dim3 g(cdivu((Nt + 15) >> 4, 8), 32 / 32);
    k_conv_wmma<2><<<g, blk, 0, stream>>>(bufB, wb2, cb2, bufA,
        512, 16, 47, 47, 32, 4, 4, 44, 44, 256, 256); }
  // pool2: 44 -> 22                                      bufA -> bufB
  { long n = 512l * 32 * 22 * 22;
    k_maxpool2_bf16<<<dim3(cdivu(n, 256)), blk, 0, stream>>>(bufA, bufB, 512 * 32, 44, 44); }

  // conv3: 32x22x22 -> 64x18x18 (K=800), MBLK=2          bufB -> bufA
  { long Nt = 512l * 18 * 18; dim3 g(cdivu((Nt + 15) >> 4, 8), 64 / 32);
    k_conv_wmma<2><<<g, blk, 0, stream>>>(bufB, wb3, cb3, bufA,
        512, 32, 22, 22, 64, 5, 5, 18, 18, 800, 800); }
  // conv4: 64x18x18 -> 128x14x14 (K=1600), MBLK=2        bufA -> bufB
  { long Nt = 512l * 14 * 14; dim3 g(cdivu((Nt + 15) >> 4, 8), 128 / 32);
    k_conv_wmma<2><<<g, blk, 0, stream>>>(bufA, wb4, cb4, bufB,
        512, 64, 18, 18, 128, 5, 5, 14, 14, 1600, 1600); }
  // conv5: 128x14x14 -> 128x10x10 (K=3200), MBLK=2       bufB -> bufA
  { long Nt = 512l * 10 * 10; dim3 g(cdivu((Nt + 15) >> 4, 8), 128 / 32);
    k_conv_wmma<2><<<g, blk, 0, stream>>>(bufB, wb5, cb5, bufA,
        512, 128, 14, 14, 128, 5, 5, 10, 10, 3200, 3200); }
  // conv6: 128x10x10 -> 128x8x8 (K=1152), MBLK=2         bufA -> bufB
  { long Nt = 512l * 8 * 8; dim3 g(cdivu((Nt + 15) >> 4, 8), 128 / 32);
    k_conv_wmma<2><<<g, blk, 0, stream>>>(bufA, wb6, cb6, bufB,
        512, 128, 10, 10, 128, 3, 3, 8, 8, 1152, 1152); }

  // fc1: [512,8192] @ lw1^T -> [512,1024] bf16           bufB -> bufA
  { dim3 g(cdivu(cdivu(512, 32), 8), 1024 / 16);
    k_fc_wmma<<<g, blk, 0, stream>>>(bufB, wf1, lb1, bufA, nullptr,
                                     1024, 8192, 512, 0); }
  // fc2: [512,1024] @ lw2^T, sigmoid -> [512,384] f32    bufA -> seq
  { dim3 g(cdivu(cdivu(512, 32), 8), 384 / 16);
    k_fc_wmma<<<g, blk, 0, stream>>>(bufA, wf2, lb2, nullptr, seq,
                                     384, 1024, 512, 1); }

  // LSTM scan over T=512, writes [512, 384] = (512, 6, 8, 8) to d_out
  k_lstm<<<dim3(1), dim3(512), 0, stream>>>(seq, wih, whh, bih, bhh, out, 512);
}
